// RelationnetBBoxNetworkEncoder_44066364457583
// MI455X (gfx1250) — matrix operand order repaired and checked by reading, not compiled, not run on hardware
//
#include <hip/hip_runtime.h>
#include <hip/hip_bf16.h>

typedef __attribute__((ext_vector_type(8)))  __bf16 v8bf;
typedef __attribute__((ext_vector_type(16))) __bf16 v16bf;
typedef __attribute__((ext_vector_type(8)))  float  v8f;

#define B_   16
#define M_   64
#define C_   1280
#define H_   1280
#define LDST 1288   // padded LDS row stride (bf16 elems); 2576B, 16B-aligned, breaks bank wrap

static __device__ __forceinline__ unsigned short f2bf(float f) {
  unsigned u = __float_as_uint(f);
  u += 0x7FFFu + ((u >> 16) & 1u);          // round-to-nearest-even
  return (unsigned short)(u >> 16);
}

static __device__ __forceinline__ v16bf make_frag(const unsigned short* p0,
                                                  const unsigned short* p1) {
  union { v16bf v; v8bf h[2]; } u;
  u.h[0] = *(const v8bf*)p0;                // 16B load (global_load_b128 / ds_load_b128)
  u.h[1] = *(const v8bf*)p1;
  return u.v;
}

#define WMMA_BF16(A, Bf, Cacc) \
  __builtin_amdgcn_wmma_f32_16x16x32_bf16(false, (A), false, (Bf), (short)0, (Cacc), false, false)

// ---------- prep kernels ----------
__global__ void k_cast_bf16(const float* __restrict__ src,
                            unsigned short* __restrict__ dst, int n) {
  int i = blockIdx.x * blockDim.x + threadIdx.x;
  if (i < n) dst[i] = f2bf(src[i]);
}

// src: R x C row-major fp32  ->  dst: C x R row-major bf16 (dst[j][i] = src[i][j])
__global__ void k_transpose_bf16(const float* __restrict__ src,
                                 unsigned short* __restrict__ dst, int R, int C) {
  int i = blockIdx.x * blockDim.x + threadIdx.x;
  if (i < R * C) {
    int j  = i / R;
    int i2 = i - j * R;
    dst[i] = f2bf(src[i2 * C + j]);
  }
}

__global__ void k_zero(float* __restrict__ p, int n) {
  int i = blockIdx.x * blockDim.x + threadIdx.x;
  if (i < n) p[i] = 0.f;
}

// ---------- layer 1: hi/hj = roi @ W1_i / W1_j ----------
// A: roi bf16 [1024][1280]; W1t: bf16 [1280][2560] with W1t[h][c2] = W1[c2][h]
// grid (10, 32): 256 cols x 32 rows per block; each wave: 2 row-tiles x 2 col-tiles
__global__ __launch_bounds__(256) void k_gemm1(const unsigned short* __restrict__ A,
                                               const unsigned short* __restrict__ W1t,
                                               float* __restrict__ hi,
                                               float* __restrict__ hj) {
  const int tid = threadIdx.x;
  const int wv = tid >> 5, lane = tid & 31;
  const int ln = lane & 15, kh = lane >> 4;
  const int r0 = blockIdx.y * 32;
  const int n0 = blockIdx.x * 256 + wv * 32;  // two adjacent 16-col tiles
  const int col0 = n0 + ln;
  const int col1 = col0 + 16;
  const int hcol0 = (col0 < H_) ? col0 : col0 - H_;
  const int hcol1 = (col1 < H_) ? col1 : col1 - H_;
  const unsigned short* brow0 = W1t + hcol0 * (2 * C_) + ((col0 < H_) ? 0 : C_);
  const unsigned short* brow1 = W1t + hcol1 * (2 * C_) + ((col1 < H_) ? 0 : C_);
  const unsigned short* arow0 = A + (r0 + ln) * C_;
  const unsigned short* arow1 = arow0 + 16 * C_;
  v8f acc00 = {}, acc01 = {}, acc10 = {}, acc11 = {};
  for (int k0 = 0; k0 < C_; k0 += 32) {
    v16bf b0 = make_frag(brow0 + k0 + kh * 16, brow0 + k0 + kh * 16 + 8);
    v16bf b1 = make_frag(brow1 + k0 + kh * 16, brow1 + k0 + kh * 16 + 8);
    v16bf a0 = make_frag(arow0 + k0 + kh * 16, arow0 + k0 + kh * 16 + 8);
    v16bf a1 = make_frag(arow1 + k0 + kh * 16, arow1 + k0 + kh * 16 + 8);
    acc00 = WMMA_BF16(a0, b0, acc00);
    acc10 = WMMA_BF16(a1, b0, acc10);
    acc01 = WMMA_BF16(a0, b1, acc01);
    acc11 = WMMA_BF16(a1, b1, acc11);
  }
  float* dst0 = (col0 < H_) ? hi : hj;
  float* dst1 = (col1 < H_) ? hi : hj;
#pragma unroll
  for (int i = 0; i < 8; ++i) {
    int row = r0 + kh * 8 + i;
    dst0[row * H_ + hcol0]        = acc00[i];
    dst0[(row + 16) * H_ + hcol0] = acc10[i];
    dst1[row * H_ + hcol1]        = acc01[i];
    dst1[(row + 16) * H_ + hcol1] = acc11[i];
  }
}

// ---------- fused layers 2+3 + reduction ----------
// grid: (mhalf=2, n=64, b=16); block 256 = 8 waves; each wave: 2 row-tiles x 2 col-tiles
__global__ __launch_bounds__(256) void k_fused23(const float* __restrict__ hi,
                                                 const float* __restrict__ hj,
                                                 const float* __restrict__ b1,
                                                 const unsigned short* __restrict__ W2t,
                                                 const float* __restrict__ b2,
                                                 const unsigned short* __restrict__ W3t,
                                                 const float* __restrict__ b3,
                                                 float* __restrict__ out) {
  extern __shared__ char smem_raw[];
  unsigned short* h1s = (unsigned short*)smem_raw;                       // 32 x LDST bf16
  unsigned short* h2s = (unsigned short*)(smem_raw + 32 * LDST * 2);     // 32 x LDST bf16
  float* partial      = (float*)(smem_raw + 2 * 32 * LDST * 2);          // 1280 f32

  const int tid = threadIdx.x;
  const int wv = tid >> 5, lane = tid & 31;
  const int ln = lane & 15, kh = lane >> 4;
  const int m0 = blockIdx.x * 32;
  const int nrow = blockIdx.y;
  const int b = blockIdx.z;

  // ---- step 1: h1 slab = relu(hi[b,m,:] + hj[b,n,:] + b1) -> LDS bf16; zero partial
  const float* hjrow = hj + (b * M_ + nrow) * H_;
  float hjv[5], b1v[5];
#pragma unroll
  for (int k = 0; k < 5; ++k) {
    int h = tid + 256 * k;
    hjv[k] = hjrow[h];
    b1v[k] = b1[h];
    partial[h] = 0.f;
  }
  const float* hibase = hi + (b * M_ + m0) * H_;
  for (int mm = 0; mm < 32; ++mm) {
    const float* hirow = hibase + mm * H_;
#pragma unroll
    for (int k = 0; k < 5; ++k) {
      int h = tid + 256 * k;
      float v = hirow[h] + hjv[k] + b1v[k];
      h1s[mm * LDST + h] = f2bf(fmaxf(v, 0.f));
    }
  }
  __syncthreads();

  // ---- step 2: h2 = relu(h1 @ W2 + b2) -> LDS bf16
  for (int t = 0; t < 5; ++t) {
    const int n0 = (2 * wv + 16 * t) * 16;    // two adjacent 16-col tiles
    const int col0 = n0 + ln, col1 = col0 + 16;
    const unsigned short* brow0 = W2t + col0 * H_;
    const unsigned short* brow1 = W2t + col1 * H_;
    if (t < 4) {                               // pull next tile's weight rows toward WGP
      __builtin_prefetch(brow0 + 256 * H_, 0, 0);
      __builtin_prefetch(brow1 + 256 * H_, 0, 0);
    }
    const unsigned short* a0p = h1s + ln * LDST;
    const unsigned short* a1p = a0p + 16 * LDST;
    v8f acc00 = {}, acc01 = {}, acc10 = {}, acc11 = {};
    for (int k0 = 0; k0 < H_; k0 += 32) {
      v16bf bf0 = make_frag(brow0 + k0 + kh * 16, brow0 + k0 + kh * 16 + 8);
      v16bf bf1 = make_frag(brow1 + k0 + kh * 16, brow1 + k0 + kh * 16 + 8);
      v16bf a0  = make_frag(a0p + k0 + kh * 16, a0p + k0 + kh * 16 + 8);
      v16bf a1  = make_frag(a1p + k0 + kh * 16, a1p + k0 + kh * 16 + 8);
      acc00 = WMMA_BF16(a0, bf0, acc00);
      acc10 = WMMA_BF16(a1, bf0, acc10);
      acc01 = WMMA_BF16(a0, bf1, acc01);
      acc11 = WMMA_BF16(a1, bf1, acc11);
    }
    const float bias0 = b2[col0];
    const float bias1 = b2[col1];
#pragma unroll
    for (int i = 0; i < 8; ++i) {
      int r = kh * 8 + i;
      h2s[r * LDST + col0]        = f2bf(fmaxf(acc00[i] + bias0, 0.f));
      h2s[(r + 16) * LDST + col0] = f2bf(fmaxf(acc10[i] + bias0, 0.f));
      h2s[r * LDST + col1]        = f2bf(fmaxf(acc01[i] + bias1, 0.f));
      h2s[(r + 16) * LDST + col1] = f2bf(fmaxf(acc11[i] + bias1, 0.f));
    }
  }
  __syncthreads();

  // ---- step 3: h3 = relu(h2 @ W3 + b3), row-reduced into LDS partial
  for (int t = 0; t < 5; ++t) {
    const int n0 = (2 * wv + 16 * t) * 16;
    const int col0 = n0 + ln, col1 = col0 + 16;
    const unsigned short* brow0 = W3t + col0 * H_;
    const unsigned short* brow1 = W3t + col1 * H_;
    if (t < 4) {
      __builtin_prefetch(brow0 + 256 * H_, 0, 0);
      __builtin_prefetch(brow1 + 256 * H_, 0, 0);
    }
    const unsigned short* a0p = h2s + ln * LDST;
    const unsigned short* a1p = a0p + 16 * LDST;
    v8f acc00 = {}, acc01 = {}, acc10 = {}, acc11 = {};
    for (int k0 = 0; k0 < H_; k0 += 32) {
      v16bf bf0 = make_frag(brow0 + k0 + kh * 16, brow0 + k0 + kh * 16 + 8);
      v16bf bf1 = make_frag(brow1 + k0 + kh * 16, brow1 + k0 + kh * 16 + 8);
      v16bf a0  = make_frag(a0p + k0 + kh * 16, a0p + k0 + kh * 16 + 8);
      v16bf a1  = make_frag(a1p + k0 + kh * 16, a1p + k0 + kh * 16 + 8);
      acc00 = WMMA_BF16(a0, bf0, acc00);
      acc10 = WMMA_BF16(a1, bf0, acc10);
      acc01 = WMMA_BF16(a0, bf1, acc01);
      acc11 = WMMA_BF16(a1, bf1, acc11);
    }
    const float bias0 = b3[col0];
    const float bias1 = b3[col1];
    float s0 = 0.f, s1 = 0.f;
#pragma unroll
    for (int i = 0; i < 8; ++i) {
      s0 += fmaxf(acc00[i] + bias0, 0.f);
      s0 += fmaxf(acc10[i] + bias0, 0.f);
      s1 += fmaxf(acc01[i] + bias1, 0.f);
      s1 += fmaxf(acc11[i] + bias1, 0.f);
    }
    atomicAdd(&partial[col0], s0);   // ds_add_f32; two lanes (kh=0/1) cover all 32 rows
    atomicAdd(&partial[col1], s1);
  }
  __syncthreads();

  // ---- step 4: accumulate into out[b,:]
  float* orow = out + b * H_;
#pragma unroll
  for (int k = 0; k < 5; ++k) {
    int h = tid + 256 * k;
    atomicAdd(&orow[h], partial[h]);
  }
}

// ---------- host launcher ----------
extern "C" void kernel_launch(void* const* d_in, const int* in_sizes, int n_in,
                              void* d_out, int out_size, void* d_ws, size_t ws_size,
                              hipStream_t stream) {
  const float* roi = (const float*)d_in[0];
  const float* W1  = (const float*)d_in[1];
  const float* b1  = (const float*)d_in[2];
  const float* W2  = (const float*)d_in[3];
  const float* b2  = (const float*)d_in[4];
  const float* W3  = (const float*)d_in[5];
  const float* b3  = (const float*)d_in[6];
  float* out = (float*)d_out;
  char* ws = (char*)d_ws;

  // workspace layout (bytes), total ~26.2 MB
  unsigned short* roi_bf = (unsigned short*)(ws + 0);          // 1310720 bf16
  unsigned short* w1t    = (unsigned short*)(ws + 2621440);    // 1280x2560 bf16
  unsigned short* w2t    = (unsigned short*)(ws + 9175040);    // 1280x1280 bf16
  unsigned short* w3t    = (unsigned short*)(ws + 12451840);   // 1280x1280 bf16
  float* hi              = (float*)(ws + 15728640);            // 1024x1280 f32
  float* hj              = (float*)(ws + 20971520);            // 1024x1280 f32

  k_cast_bf16<<<(1310720 + 255) / 256, 256, 0, stream>>>(roi, roi_bf, 1310720);
  k_transpose_bf16<<<(3276800 + 255) / 256, 256, 0, stream>>>(W1, w1t, 2560, 1280);
  k_transpose_bf16<<<(1638400 + 255) / 256, 256, 0, stream>>>(W2, w2t, 1280, 1280);
  k_transpose_bf16<<<(1638400 + 255) / 256, 256, 0, stream>>>(W3, w3t, 1280, 1280);
  k_zero<<<(B_ * H_ + 255) / 256, 256, 0, stream>>>(out, B_ * H_);

  k_gemm1<<<dim3(10, 32), 256, 0, stream>>>(roi_bf, w1t, hi, hj);

  size_t smem = (size_t)(2 * 32 * LDST * 2 + H_ * 4);  // 169,984 B (< 320 KB WGP LDS)
  hipFuncSetAttribute((const void*)k_fused23,
                      hipFuncAttributeMaxDynamicSharedMemorySize, (int)smem);
  k_fused23<<<dim3(2, 64, 16), 256, smem, stream>>>(hi, hj, b1, w2t, b2, w3t, b3, out);
}
